// RWLayer_7189775253599
// MI455X (gfx1250) — matrix operand order repaired and checked by reading, not compiled, not run on hardware
//
#include <hip/hip_runtime.h>
#include <math.h>

// ---------------------------------------------------------------------------
// RWLayer: out = softmax(normalize(x) @ normalize(x)^T) @ x
// N = 8192 rows, D = 2048 cols, fp32 in/out.
// Cosine logits are bounded in [-1,1] -> exp() needs no max-subtraction, so
// softmax == exp(s)/rowsum(exp(s)) exactly. Pipeline (ws ~= 160 MB):
//   1) Qh = f16(x / max(||x_row||, eps))                  [ws +0,    32 MB]
//   2) E  = f16(exp(Qh @ Qh^T))   (WMMA GEMM, fused exp)  [ws +32M, 128 MB]
//   3) linv[i] = 1 / sum_j E[i][j]                        [ws +160M, 32 KB]
//   4) out = (E @ f16(x)) * linv[row]   (WMMA GEMM)
// GEMM kernels: 256x128 block tile, BK=32, 8 waves in 4x2; each wave owns a
// 64x64 patch = 4x4 v_wmma_f32_16x16x32_f16 tiles (16 ds_load_b128 per 16
// WMMA). Double-buffered LDS (one barrier per K-step) + register prefetch.
// ---------------------------------------------------------------------------

typedef __attribute__((ext_vector_type(16))) _Float16 v16h;
typedef __attribute__((ext_vector_type(8)))  _Float16 v8h;
typedef __attribute__((ext_vector_type(4)))  _Float16 v4h;
typedef __attribute__((ext_vector_type(8)))  float    v8f;

#define NROWS 8192
#define NDIM  2048
#define LDT   40   // padded LDS row stride in halfs (80 B, keeps 16B alignment)

static __device__ __forceinline__ v16h make_v16(v8h lo, v8h hi) {
  union { v16h v; v8h h[2]; } u;
  u.h[0] = lo;
  u.h[1] = hi;
  return u.v;
}

// ---------------------------------------------------------------------------
// Kernel 1: per-row L2 normalize, emit f16 normalized rows.
// ---------------------------------------------------------------------------
__global__ __launch_bounds__(256) void normalize_kernel(const float* __restrict__ x,
                                                        _Float16* __restrict__ Q) {
  const int row = blockIdx.x;
  const int t   = threadIdx.x;
  const float4* x4 = (const float4*)(x + (size_t)row * NDIM);
  float4 a = x4[t];
  float4 b = x4[t + 256];
  float ss = a.x * a.x + a.y * a.y + a.z * a.z + a.w * a.w +
             b.x * b.x + b.y * b.y + b.z * b.z + b.w * b.w;
  __shared__ float red[256];
  red[t] = ss;
  __syncthreads();
  for (int off = 128; off > 0; off >>= 1) {
    if (t < off) red[t] += red[t + off];
    __syncthreads();
  }
  const float inv = 1.0f / fmaxf(sqrtf(red[0]), 1e-12f);
  v4h qa = { (_Float16)(a.x * inv), (_Float16)(a.y * inv),
             (_Float16)(a.z * inv), (_Float16)(a.w * inv) };
  v4h qb = { (_Float16)(b.x * inv), (_Float16)(b.y * inv),
             (_Float16)(b.z * inv), (_Float16)(b.w * inv) };
  _Float16* qr = Q + (size_t)row * NDIM;
  *(v4h*)(qr + t * 4)        = qa;
  *(v4h*)(qr + 1024 + t * 4) = qb;
}

// ---------------------------------------------------------------------------
// Kernel 2: E = f16(exp(Q @ Q^T)).
// ---------------------------------------------------------------------------
__global__ __launch_bounds__(256) void gemm1_exp_kernel(const _Float16* __restrict__ Q,
                                                        _Float16* __restrict__ E) {
  __shared__ _Float16 As[2 * 256 * LDT];   // 40 KB
  __shared__ _Float16 Bs[2 * 128 * LDT];   // 20 KB
  const int t    = threadIdx.x;
  const int lane = t & 31;
  const int w    = t >> 5;
  const int wm   = w & 3;   // 4 waves along M (64 rows each)
  const int wn   = w >> 2;  // 2 waves along N (64 cols each)
  const int bm   = blockIdx.y * 256;
  const int bn   = blockIdx.x * 128;

  // A loader: one 32-half row per thread; B loader: 2 threads per row.
  const _Float16* gA = Q + (size_t)(bm + t) * NDIM;
  const int lrB = t >> 1;
  const int lcB = (t & 1) << 4;
  const _Float16* gB = Q + (size_t)(bn + lrB) * NDIM + lcB;

  v8h a[4], b[2];
  auto gload = [&](int kb) {
    const _Float16* pA = gA + kb * 32;
    a[0] = *(const v8h*)(pA);
    a[1] = *(const v8h*)(pA + 8);
    a[2] = *(const v8h*)(pA + 16);
    a[3] = *(const v8h*)(pA + 24);
    const _Float16* pB = gB + kb * 32;
    b[0] = *(const v8h*)(pB);
    b[1] = *(const v8h*)(pB + 8);
  };
  auto sstore = [&](int buf) {
    _Float16* sA = As + buf * (256 * LDT) + t * LDT;
    *(v8h*)(sA)      = a[0];
    *(v8h*)(sA + 8)  = a[1];
    *(v8h*)(sA + 16) = a[2];
    *(v8h*)(sA + 24) = a[3];
    _Float16* sB = Bs + buf * (128 * LDT) + lrB * LDT + lcB;
    *(v8h*)(sB)     = b[0];
    *(v8h*)(sB + 8) = b[1];
  };

  const v8f zero = {0.f, 0.f, 0.f, 0.f, 0.f, 0.f, 0.f, 0.f};
  v8f acc[4][4];
#pragma unroll
  for (int mt = 0; mt < 4; ++mt)
#pragma unroll
    for (int nt = 0; nt < 4; ++nt) acc[mt][nt] = zero;

  const int half = lane >> 4;  // which K-half of the fragment this lane holds
  const int l16  = lane & 15;
  const int KT   = NDIM / 32;  // 64

  gload(0);
  sstore(0);
  gload(1);
  __syncthreads();

  for (int kb = 0; kb < KT; ++kb) {
    const int cur = kb & 1;
    if (kb + 1 < KT) sstore(cur ^ 1);   // stage tile kb+1 into the other buffer
    if (kb + 2 < KT) gload(kb + 2);     // prefetch tile kb+2 into registers

    const _Float16* sA = As + cur * (256 * LDT);
    const _Float16* sB = Bs + cur * (128 * LDT);
    v16h bf[4];
#pragma unroll
    for (int nt = 0; nt < 4; ++nt) {
      const int col = wn * 64 + nt * 16 + l16;
      // B k order used here (contiguous, matches A): lanes 0-15 hold K=0..15, lanes 16-31 hold K=16..31
      v8h lo = *(const v8h*)&sB[col * LDT + half * 16];
      v8h hi = *(const v8h*)&sB[col * LDT + half * 16 + 8];
      bf[nt] = make_v16(lo, hi);
    }
#pragma unroll
    for (int mt = 0; mt < 4; ++mt) {
      const int row = wm * 64 + mt * 16 + l16;
      // A k order used here (contiguous, matches B): lanes 0-15 hold K=0..15, lanes 16-31 hold K=16..31
      v8h lo = *(const v8h*)&sA[row * LDT + half * 16];
      v8h hi = *(const v8h*)&sA[row * LDT + half * 16 + 8];
      v16h af = make_v16(lo, hi);
#pragma unroll
      for (int nt = 0; nt < 4; ++nt)
        acc[mt][nt] = __builtin_amdgcn_wmma_f32_16x16x32_f16(
            false, af, false, bf[nt], (short)0, acc[mt][nt], false, false);
    }
    __syncthreads();
  }

  // fused exp epilogue; logits in [-1,1] so exp is bounded by e
#pragma unroll
  for (int mt = 0; mt < 4; ++mt) {
    const int row0 = bm + wm * 64 + mt * 16 + half * 8;
#pragma unroll
    for (int nt = 0; nt < 4; ++nt) {
      const int col = bn + wn * 64 + nt * 16 + l16;
#pragma unroll
      for (int r = 0; r < 8; ++r) {
        E[(size_t)(row0 + r) * NROWS + col] = (_Float16)__expf(acc[mt][nt][r]);
      }
    }
  }
}

// ---------------------------------------------------------------------------
// Kernel 3: linv[row] = 1 / sum_j E[row][j]
// ---------------------------------------------------------------------------
__global__ __launch_bounds__(256) void rowsum_kernel(const _Float16* __restrict__ E,
                                                     float* __restrict__ linv) {
  const int row = blockIdx.x;
  const int t   = threadIdx.x;
  const _Float16* er = E + (size_t)row * NROWS;
  float s = 0.f;
#pragma unroll
  for (int i = 0; i < 4; ++i) {
    v8h v = *(const v8h*)(er + (size_t)i * 2048 + t * 8);
#pragma unroll
    for (int j = 0; j < 8; ++j) s += (float)v[j];
  }
  __shared__ float red[256];
  red[t] = s;
  __syncthreads();
  for (int off = 128; off > 0; off >>= 1) {
    if (t < off) red[t] += red[t + off];
    __syncthreads();
  }
  if (t == 0) linv[row] = 1.0f / red[0];
}

// ---------------------------------------------------------------------------
// Kernel 4: out = (E @ f16(x)) * linv[row].  B tile (32x128 of x, fp32) is
// converted to f16 and transposed into LDS as [n][k].
// ---------------------------------------------------------------------------
__global__ __launch_bounds__(256) void gemm2_kernel(const _Float16* __restrict__ E,
                                                    const float* __restrict__ x,
                                                    const float* __restrict__ linv,
                                                    float* __restrict__ out) {
  __shared__ _Float16 As[2 * 256 * LDT];
  __shared__ _Float16 Bs[2 * 128 * LDT];  // [n][k] layout, stride LDT
  const int t    = threadIdx.x;
  const int lane = t & 31;
  const int w    = t >> 5;
  const int wm   = w & 3;
  const int wn   = w >> 2;
  const int bm   = blockIdx.y * 256;
  const int bn   = blockIdx.x * 128;

  const _Float16* gA = E + (size_t)(bm + t) * NROWS;
  const int kr = t >> 3;        // 0..31 (k row within tile)
  const int nc = (t & 7) << 4;  // 0..112 (n offset, 16 floats)
  const float* gB = x + (size_t)kr * NDIM + bn + nc;

  v8h a[4];
  float4 f0, f1, f2, f3;
  auto gload = [&](int kb) {
    const _Float16* pA = gA + kb * 32;
    a[0] = *(const v8h*)(pA);
    a[1] = *(const v8h*)(pA + 8);
    a[2] = *(const v8h*)(pA + 16);
    a[3] = *(const v8h*)(pA + 24);
    const float* pB = gB + (size_t)kb * 32 * NDIM;
    f0 = *(const float4*)(pB);
    f1 = *(const float4*)(pB + 4);
    f2 = *(const float4*)(pB + 8);
    f3 = *(const float4*)(pB + 12);
  };
  auto sstore = [&](int buf) {
    _Float16* sA = As + buf * (256 * LDT) + t * LDT;
    *(v8h*)(sA)      = a[0];
    *(v8h*)(sA + 8)  = a[1];
    *(v8h*)(sA + 16) = a[2];
    *(v8h*)(sA + 24) = a[3];
    _Float16* sB = Bs + buf * (128 * LDT);
    const float fb[16] = {f0.x, f0.y, f0.z, f0.w, f1.x, f1.y, f1.z, f1.w,
                          f2.x, f2.y, f2.z, f2.w, f3.x, f3.y, f3.z, f3.w};
#pragma unroll
    for (int j = 0; j < 16; ++j)
      sB[(nc + j) * LDT + kr] = (_Float16)fb[j];  // transpose into [n][k]
  };

  const v8f zero = {0.f, 0.f, 0.f, 0.f, 0.f, 0.f, 0.f, 0.f};
  v8f acc[4][4];
#pragma unroll
  for (int mt = 0; mt < 4; ++mt)
#pragma unroll
    for (int nt = 0; nt < 4; ++nt) acc[mt][nt] = zero;

  const int half = lane >> 4;
  const int l16  = lane & 15;
  const int KT   = NROWS / 32;  // 256

  gload(0);
  sstore(0);
  gload(1);
  __syncthreads();

  for (int kb = 0; kb < KT; ++kb) {
    const int cur = kb & 1;
    if (kb + 1 < KT) sstore(cur ^ 1);
    if (kb + 2 < KT) gload(kb + 2);

    const _Float16* sA = As + cur * (256 * LDT);
    const _Float16* sB = Bs + cur * (128 * LDT);
    v16h bf[4];
#pragma unroll
    for (int nt = 0; nt < 4; ++nt) {
      const int col = wn * 64 + nt * 16 + l16;
      v8h lo = *(const v8h*)&sB[col * LDT + half * 16];
      v8h hi = *(const v8h*)&sB[col * LDT + half * 16 + 8];
      bf[nt] = make_v16(lo, hi);
    }
#pragma unroll
    for (int mt = 0; mt < 4; ++mt) {
      const int row = wm * 64 + mt * 16 + l16;
      v8h lo = *(const v8h*)&sA[row * LDT + half * 16];
      v8h hi = *(const v8h*)&sA[row * LDT + half * 16 + 8];
      v16h af = make_v16(lo, hi);
#pragma unroll
      for (int nt = 0; nt < 4; ++nt)
        acc[mt][nt] = __builtin_amdgcn_wmma_f32_16x16x32_f16(
            false, af, false, bf[nt], (short)0, acc[mt][nt], false, false);
    }
    __syncthreads();
  }

  // scale by 1/rowsum and store fp32
#pragma unroll
  for (int mt = 0; mt < 4; ++mt) {
    const int row0 = bm + wm * 64 + mt * 16 + half * 8;
#pragma unroll
    for (int r = 0; r < 8; ++r) {
      const float s = linv[row0 + r];
#pragma unroll
      for (int nt = 0; nt < 4; ++nt) {
        const int col = bn + wn * 64 + nt * 16 + l16;
        out[(size_t)(row0 + r) * NDIM + col] = acc[mt][nt][r] * s;
      }
    }
  }
}

// ---------------------------------------------------------------------------
extern "C" void kernel_launch(void* const* d_in, const int* in_sizes, int n_in,
                              void* d_out, int out_size, void* d_ws, size_t ws_size,
                              hipStream_t stream) {
  (void)in_sizes; (void)n_in; (void)out_size; (void)ws_size;
  const float* x   = (const float*)d_in[0];
  float*       out = (float*)d_out;
  char*        ws  = (char*)d_ws;

  _Float16* Qh   = (_Float16*)ws;                                       // 32 MB
  _Float16* E    = (_Float16*)(ws + (size_t)32 * 1024 * 1024);          // 128 MB
  float*    linv = (float*)(ws + (size_t)160 * 1024 * 1024);            // 32 KB

  normalize_kernel<<<NROWS, 256, 0, stream>>>(x, Qh);

  dim3 g1(NROWS / 128, NROWS / 256);
  gemm1_exp_kernel<<<g1, 256, 0, stream>>>(Qh, E);

  rowsum_kernel<<<NROWS, 256, 0, stream>>>(E, linv);

  dim3 g2(NDIM / 128, NROWS / 256);
  gemm2_kernel<<<g2, 256, 0, stream>>>(E, x, linv, out);
}
